// AffordancePredictor_72009421685020
// MI455X (gfx1250) — matrix operand order repaired and checked by reading, not compiled, not run on hardware
//
#include <hip/hip_runtime.h>
#include <hip/hip_bf16.h>

typedef __attribute__((ext_vector_type(16))) _Float16 v16h;
typedef __attribute__((ext_vector_type(8)))  _Float16 v8h;
typedef __attribute__((ext_vector_type(8)))  float    v8f;

__device__ __forceinline__ v16h ld_fragA(const _Float16* p) {
    v8h x = *(const v8h*)p;         // K 0..7   (or 8..15 for hi lanes)
    v8h y = *(const v8h*)(p + 16);  // K 16..23 (or 24..31)
    return __builtin_shufflevector(x, y, 0,1,2,3,4,5,6,7,8,9,10,11,12,13,14,15);
}
__device__ __forceinline__ v16h ld_fragB(const _Float16* p) {
    v8h x = *(const v8h*)p;
    v8h y = *(const v8h*)(p + 16);
    return __builtin_shufflevector(x, y, 0,1,2,3,4,5,6,7,8,9,10,11,12,13,14,15);
}
__device__ __forceinline__ int iclamp(int x, int lo, int hi) {
    return x < lo ? lo : (x > hi ? hi : x);
}

// ---------------------------------------------------------------------------
// BN fold:  scale = g * rsqrt(v + eps) ; shift = b - m * scale
// ---------------------------------------------------------------------------
__global__ void bn_prep_kernel(const float* __restrict__ g,
                               const float* __restrict__ b,
                               const float* __restrict__ m,
                               const float* __restrict__ v,
                               float* __restrict__ scale,
                               float* __restrict__ shift, int C) {
    int i = blockIdx.x * blockDim.x + threadIdx.x;
    if (i < C) {
        float s = g[i] * rsqrtf(v[i] + 1e-5f);
        scale[i] = s;
        shift[i] = b[i] - m[i] * s;
    }
}

// ---------------------------------------------------------------------------
// fp32 -> f16 image conversion
// ---------------------------------------------------------------------------
__global__ void f32_to_f16_kernel(const float* __restrict__ in,
                                  _Float16* __restrict__ out, int n) {
    int i = blockIdx.x * blockDim.x + threadIdx.x;
    if (i < n) out[i] = (_Float16)in[i];
}

// ---------------------------------------------------------------------------
// Implicit-GEMM conv + folded BN (+ optional residual) (+ optional ReLU)
//   M tile 64 (4 sub-tiles/wave), N tile 128, K = Cin*KHW^2 in chunks of 32.
//   Double-buffered LDS panels: chunk c+1's global loads are issued into
//   registers while chunk c's 4 WMMAs run; one barrier per chunk.
// ---------------------------------------------------------------------------
template <int KHW>
__global__ __launch_bounds__(256) void conv_bn_wmma_kernel(
    const _Float16* __restrict__ in, const float* __restrict__ w,
    const float* __restrict__ scale, const float* __restrict__ shift,
    const _Float16* __restrict__ residual, _Float16* __restrict__ out,
    int B, int Cin, int Hin, int Win,
    int Cout, int Hout, int Wout,
    int strd, int pad, int do_relu)
{
    constexpr int KK = KHW * KHW;
    const int lane = threadIdx.x & 31;
    const int wv   = threadIdx.x >> 5;            // 0..7
    const int n_tile  = blockIdx.x * 128;
    const int co_base = blockIdx.y * 64;
    const int Ntot = B * Hout * Wout;
    const int K    = Cin * KK;
    const int HWo  = Hout * Wout;
    const int HWi  = Hin * Win;

    __shared__ __align__(16) _Float16 a_lds[2][64 * 40];    // 64 M x 32 K
    __shared__ __align__(16) _Float16 b_lds[2][128 * 40];   // 128 N x 32 K

    // ---- K-invariant per-thread staging geometry ----
    const int kk  = threadIdx.x & 31;        // same K slot for every panel row
    const int row = threadIdx.x >> 5;        // 0..7
    int nbase[16], ih0[16], iw0[16];
    unsigned validn = 0;
    #pragma unroll
    for (int i = 0; i < 16; ++i) {
        int ng = n_tile + row + i * 8;
        bool vn = ng < Ntot;
        int ngc = vn ? ng : 0;
        int b  = ngc / HWo;
        int sr = ngc - b * HWo;
        int oh = sr / Wout;
        int ow = sr - oh * Wout;
        nbase[i] = b * Cin * HWi;
        ih0[i]   = oh * strd - pad;
        iw0[i]   = ow * strd - pad;
        if (vn) validn |= (1u << i);
    }

    // staging registers (shared between issue/commit, SROA keeps in VGPRs)
    float    wreg[8];
    _Float16 breg[16];
    unsigned pm = 0;
    bool     vk = false;

    auto issue = [&](int k0) {
        int  kg  = k0 + kk;
        vk       = kg < K;
        int  kgc = vk ? kg : K - 1;
        int  cin = kgc / KK;                  // constant divisor
        int  rem = kgc - cin * KK;
        int  kh  = rem / KHW;
        int  kw  = rem - kh * KHW;
        int  coff = cin * HWi;
        #pragma unroll
        for (int i = 0; i < 8; ++i)
            wreg[i] = w[(size_t)(co_base + row + i * 8) * K + kgc];
        pm = 0;
        #pragma unroll
        for (int i = 0; i < 16; ++i) {
            int ih = ih0[i] + kh;
            int iw = iw0[i] + kw;
            bool p = vk && ((validn >> i) & 1u) &&
                     (unsigned)ih < (unsigned)Hin && (unsigned)iw < (unsigned)Win;
            int ihc = iclamp(ih, 0, Hin - 1);
            int iwc = iclamp(iw, 0, Win - 1);
            breg[i] = in[(size_t)nbase[i] + coff + ihc * Win + iwc];
            if (p) pm |= (1u << i);
        }
    };
    auto commit = [&](int buf) {
        #pragma unroll
        for (int i = 0; i < 8; ++i)
            a_lds[buf][(row + i * 8) * 40 + kk] = (_Float16)(vk ? wreg[i] : 0.0f);
        #pragma unroll
        for (int i = 0; i < 16; ++i)
            b_lds[buf][(row + i * 8) * 40 + kk] =
                ((pm >> i) & 1u) ? breg[i] : (_Float16)0.0f;
    };

    v8f acc[4] = {};
    const int nchunks = (K + 31) >> 5;

    issue(0);
    commit(0);
    __syncthreads();

    const int mrow = lane & 15;
    const int hi   = lane >> 4;
    const int ncol = wv * 16 + mrow;

    for (int c = 0; c < nchunks; ++c) {
        const int cur = c & 1;
        if (c + 1 < nchunks) issue(32 * (c + 1));   // overlap with WMMAs below

        // prefetch all fragments, then fire 4 back-to-back WMMAs
        v16h bv  = ld_fragB(&b_lds[cur][ncol * 40 + hi * 8]);
        v16h av0 = ld_fragA(&a_lds[cur][( 0 + mrow) * 40 + hi * 8]);
        v16h av1 = ld_fragA(&a_lds[cur][(16 + mrow) * 40 + hi * 8]);
        v16h av2 = ld_fragA(&a_lds[cur][(32 + mrow) * 40 + hi * 8]);
        v16h av3 = ld_fragA(&a_lds[cur][(48 + mrow) * 40 + hi * 8]);
        acc[0] = __builtin_amdgcn_wmma_f32_16x16x32_f16(false, av0, false, bv, (short)0, acc[0], false, false);
        acc[1] = __builtin_amdgcn_wmma_f32_16x16x32_f16(false, av1, false, bv, (short)0, acc[1], false, false);
        acc[2] = __builtin_amdgcn_wmma_f32_16x16x32_f16(false, av2, false, bv, (short)0, acc[2], false, false);
        acc[3] = __builtin_amdgcn_wmma_f32_16x16x32_f16(false, av3, false, bv, (short)0, acc[3], false, false);

        if (c + 1 < nchunks) {
            commit(cur ^ 1);
            __syncthreads();
        }
    }

    // ---- epilogue: BN fold + residual + ReLU, f16 NCHW store ----
    const int ng = n_tile + ncol;
    if (ng < Ntot) {
        int b  = ng / HWo;
        int sr = ng - b * HWo;
        const int mbase = hi * 8;
        size_t sbase = (size_t)b * Cout * HWo + sr;
        #pragma unroll
        for (int mt = 0; mt < 4; ++mt) {
            #pragma unroll
            for (int r = 0; r < 8; ++r) {
                int co = co_base + mt * 16 + mbase + r;
                float vv = acc[mt][r] * scale[co] + shift[co];
                size_t oidx = sbase + (size_t)co * HWo;
                if (residual) vv += (float)residual[oidx];
                if (do_relu) vv = fmaxf(vv, 0.0f);
                out[oidx] = (_Float16)vv;
            }
        }
    }
}

// ---------------------------------------------------------------------------
// maxpool 3x3 stride2 pad1 (f16 in/out, f32 compare)
// ---------------------------------------------------------------------------
__global__ void maxpool_kernel(const _Float16* __restrict__ in,
                               _Float16* __restrict__ out,
                               int B, int C, int Hin, int Win, int Hout, int Wout) {
    int idx = blockIdx.x * blockDim.x + threadIdx.x;
    int total = B * C * Hout * Wout;
    if (idx >= total) return;
    int ow = idx % Wout; int t = idx / Wout;
    int oh = t % Hout;  t /= Hout;
    int c  = t % C;     int b = t / C;
    const _Float16* p = in + ((size_t)b * C + c) * Hin * Win;
    float mx = -INFINITY;
    int h0 = oh * 2 - 1, w0 = ow * 2 - 1;
    #pragma unroll
    for (int kh = 0; kh < 3; ++kh) {
        #pragma unroll
        for (int kw = 0; kw < 3; ++kw) {
            int ih = h0 + kh, iw = w0 + kw;
            if ((unsigned)ih < (unsigned)Hin && (unsigned)iw < (unsigned)Win)
                mx = fmaxf(mx, (float)p[ih * Win + iw]);
        }
    }
    out[idx] = (_Float16)mx;
}

// ---------------------------------------------------------------------------
// global average pool (f16 in, f32 out)
// ---------------------------------------------------------------------------
__global__ void gap_kernel(const _Float16* __restrict__ in, float* __restrict__ out,
                           int BC, int HW) {
    int idx = blockIdx.x * blockDim.x + threadIdx.x;
    if (idx >= BC) return;
    const _Float16* p = in + (size_t)idx * HW;
    float s = 0.0f;
    for (int i = 0; i < HW; ++i) s += (float)p[i];
    out[idx] = s / (float)HW;
}

// ---------------------------------------------------------------------------
// head: h = relu(f @ W1[e] + b1[e]) ; out = h @ W2 + b2
// ---------------------------------------------------------------------------
__global__ __launch_bounds__(64) void head_kernel(
    const float* __restrict__ feat, const int* __restrict__ cmd,
    const float* __restrict__ W1, const float* __restrict__ b1,
    const float* __restrict__ W2, const float* __restrict__ b2,
    float* __restrict__ out, int odim, int cond)
{
    int b = blockIdx.x, j = threadIdx.x;
    __shared__ float f[512];
    __shared__ float h[64];
    for (int d = j; d < 512; d += 64) f[d] = feat[(size_t)b * 512 + d];
    __syncthreads();
    int e = cond ? cmd[b] : 0;
    const float* w1 = W1 + (size_t)e * 512 * 64;
    float s = b1[e * 64 + j];
    for (int d = 0; d < 512; ++d) s += f[d] * w1[d * 64 + j];
    h[j] = fmaxf(s, 0.0f);
    __syncthreads();
    if (j < odim) {
        float o = b2[j];
        for (int k = 0; k < 64; ++k) o += h[k] * W2[k * odim + j];
        out[b * odim + j] = o;
    }
}

// ---------------------------------------------------------------------------
// host orchestration
// ---------------------------------------------------------------------------
static inline void run_conv(hipStream_t s,
                            const _Float16* in, const float* w,
                            const float* sc, const float* sh,
                            const _Float16* res, _Float16* out,
                            int B, int Cin, int Hin, int Win,
                            int Cout, int Hout, int Wout,
                            int k, int st, int pad, int relu) {
    int Ntot = B * Hout * Wout;
    dim3 grid((Ntot + 127) / 128, Cout / 64);
    if (k == 7)
        conv_bn_wmma_kernel<7><<<grid, 256, 0, s>>>(in, w, sc, sh, res, out,
            B, Cin, Hin, Win, Cout, Hout, Wout, st, pad, relu);
    else if (k == 3)
        conv_bn_wmma_kernel<3><<<grid, 256, 0, s>>>(in, w, sc, sh, res, out,
            B, Cin, Hin, Win, Cout, Hout, Wout, st, pad, relu);
    else
        conv_bn_wmma_kernel<1><<<grid, 256, 0, s>>>(in, w, sc, sh, res, out,
            B, Cin, Hin, Win, Cout, Hout, Wout, st, pad, relu);
}

struct Blk { int bn1, bn2, c1, c2, dbn, dcv, slot1, slot2, slotd; };
// d_in leaf indices from jax tree flattening (dict keys sorted)
static const Blk BLKS[8] = {
    {  7, 11, 15, 16, -1, -1,  1,  2, -1},
    { 17, 21, 25, 26, -1, -1,  3,  4, -1},
    { 27, 31, 35, 36, 37, 41,  5,  6,  7},
    { 42, 46, 50, 51, -1, -1,  8,  9, -1},
    { 52, 56, 60, 61, 62, 66, 10, 11, 12},
    { 67, 71, 75, 76, -1, -1, 13, 14, -1},
    { 77, 81, 85, 86, 87, 91, 15, 16, 17},
    { 92, 96,100,101, -1, -1, 18, 19, -1},
};

extern "C" void kernel_launch(void* const* d_in, const int* in_sizes, int n_in,
                              void* d_out, int out_size, void* d_ws, size_t ws_size,
                              hipStream_t stream) {
    (void)in_sizes; (void)n_in; (void)out_size; (void)ws_size;
    const int B = 64;
    const float* img = (const float*)d_in[0];
    const int*   cmd = (const int*)d_in[1];
    auto F = [&](int i) -> const float* { return (const float*)d_in[i]; };

    char* ws = (char*)d_ws;
    const size_t MB = 1ull << 20;
    float*    bnbuf = (float*)ws;                  // 20 slots x 1024 floats
    float*    feat  = (float*)(ws + 1 * MB);       // 64 x 512 f32
    _Float16* img16 = (_Float16*)(ws + 2 * MB);    // 19.3 MB
    _Float16* actA  = (_Float16*)(ws + 24 * MB);   // conv1 out: 103 MB
    _Float16* act1  = (_Float16*)(ws + 128 * MB);  // 26 MB each below
    _Float16* act2  = (_Float16*)(ws + 156 * MB);
    _Float16* act3  = (_Float16*)(ws + 184 * MB);
    _Float16* act4  = (_Float16*)(ws + 212 * MB);

    auto bn_slot_s = [&](int slot) { return bnbuf + (size_t)slot * 1024; };
    auto bn_slot_b = [&](int slot) { return bnbuf + (size_t)slot * 1024 + 512; };
    auto prep_bn = [&](int base_idx, int slot, int C) {
        // flattened bn dict order: b, g, m, v
        bn_prep_kernel<<<(C + 255) / 256, 256, 0, stream>>>(
            F(base_idx + 1), F(base_idx + 0), F(base_idx + 2), F(base_idx + 3),
            bn_slot_s(slot), bn_slot_b(slot), C);
    };

    // ---- fold all BN params ----
    static const int CH[4] = {64, 128, 256, 512};
    prep_bn(2, 0, 64);
    for (int bi = 0; bi < 8; ++bi) {
        int c = CH[bi / 2];
        prep_bn(BLKS[bi].bn1, BLKS[bi].slot1, c);
        prep_bn(BLKS[bi].bn2, BLKS[bi].slot2, c);
        if (BLKS[bi].dbn >= 0) prep_bn(BLKS[bi].dbn, BLKS[bi].slotd, c);
    }

    // ---- image to f16 ----
    {
        int n = B * 3 * 224 * 224;
        f32_to_f16_kernel<<<(n + 255) / 256, 256, 0, stream>>>(img, img16, n);
    }

    // ---- stem: conv7x7/s2 + BN + ReLU, then maxpool 3x3/s2 ----
    run_conv(stream, img16, F(6), bn_slot_s(0), bn_slot_b(0), nullptr, actA,
             B, 3, 224, 224, 64, 112, 112, 7, 2, 3, 1);
    {
        int total = B * 64 * 56 * 56;
        maxpool_kernel<<<(total + 255) / 256, 256, 0, stream>>>(
            actA, act1, B, 64, 112, 112, 56, 56);
    }

    // ---- residual stages ----
    _Float16* x  = act1;
    _Float16* t0 = act2;
    _Float16* t1 = act3;
    _Float16* ds = act4;
    int cin = 64, H = 56, W = 56;
    for (int s = 0; s < 4; ++s) {
        int cout = CH[s];
        for (int blk = 0; blk < 2; ++blk) {
            const Blk& bb = BLKS[s * 2 + blk];
            int st = (blk == 0 && s > 0) ? 2 : 1;
            int Ho = (st == 2) ? H / 2 : H;
            int Wo = (st == 2) ? W / 2 : W;
            run_conv(stream, x, F(bb.c1), bn_slot_s(bb.slot1), bn_slot_b(bb.slot1),
                     nullptr, t0, B, cin, H, W, cout, Ho, Wo, 3, st, 1, 1);
            const _Float16* res = x;
            if (bb.dcv >= 0) {
                run_conv(stream, x, F(bb.dcv), bn_slot_s(bb.slotd), bn_slot_b(bb.slotd),
                         nullptr, ds, B, cin, H, W, cout, Ho, Wo, 1, st, 0, 0);
                res = ds;
            }
            run_conv(stream, t0, F(bb.c2), bn_slot_s(bb.slot2), bn_slot_b(bb.slot2),
                     res, t1, B, cout, Ho, Wo, cout, Ho, Wo, 3, 1, 1, 1);
            _Float16* oldx = x; x = t1; t1 = t0; t0 = oldx;
            cin = cout; H = Ho; W = Wo;
        }
    }

    // ---- global average pool ----
    gap_kernel<<<(B * 512 + 255) / 256, 256, 0, stream>>>(x, feat, B * 512, H * W);

    // ---- heads ----
    float* out = (float*)d_out;
    head_kernel<<<B, 64, 0, stream>>>(feat, cmd, F(102), F(104), F(103), F(105),
                                      out + 0,   1, 1);   // lane
    head_kernel<<<B, 64, 0, stream>>>(feat, cmd, F(106), F(108), F(107), F(109),
                                      out + 64,  1, 1);   // route
    head_kernel<<<B, 64, 0, stream>>>(feat, cmd, F(110), F(112), F(111), F(113),
                                      out + 128, 1, 0);   // tld
    head_kernel<<<B, 64, 0, stream>>>(feat, cmd, F(114), F(116), F(115), F(117),
                                      out + 192, 2, 0);   // tls
}
